// my_gcnn_40785009443054
// MI455X (gfx1250) — matrix operand order repaired and checked by reading, not compiled, not run on hardware
//
#include <hip/hip_runtime.h>
#include <hip/hip_bf16.h>

// N=8192 nodes, DIN=DOUT=256.
// out[i,c] = relu(d_i * ((A @ hs)[i,c] + hs[i,c])),  hs[j,c] = d_j * (x@W^T)[j,c]
// d_i = rsqrt(1 + #{j : A[i,j] > 1e-15})
// d_out = [ out (8192*256 f32) | A copy (8192*8192 f32) ]

#define NN 8192
#define DD 256

typedef __attribute__((ext_vector_type(4)))  float  f4;
typedef __attribute__((ext_vector_type(8)))  float  v8f;
typedef __attribute__((ext_vector_type(16))) __bf16 v16bf;

// ---------------------------------------------------------------------------
// Kernel 1: fused degree computation + A copy to d_out (single pass over A).
// Copy stores are non-temporal: write-once data must not rinse L2.
// ---------------------------------------------------------------------------
__global__ __launch_bounds__(256) void degree_copy(const float* __restrict__ A,
                                                   float* __restrict__ Aout,
                                                   float* __restrict__ dvec) {
  const int row = blockIdx.x;
  const float* arow = A + (size_t)row * NN;
  float* orow = Aout + (size_t)row * NN;

  float s = 0.0f;
  for (int c = threadIdx.x * 4; c < NN; c += 256 * 4) {
    f4 v = *(const f4*)(arow + c);
    __builtin_nontemporal_store(v, (f4*)(orow + c));
    s += (v[0] > 1e-15f ? 1.0f : 0.0f);
    s += (v[1] > 1e-15f ? 1.0f : 0.0f);
    s += (v[2] > 1e-15f ? 1.0f : 0.0f);
    s += (v[3] > 1e-15f ? 1.0f : 0.0f);
  }
  // wave32 reduce
  for (int off = 16; off > 0; off >>= 1) s += __shfl_down(s, off, 32);
  __shared__ float red[8];
  const int wv = threadIdx.x >> 5, ln = threadIdx.x & 31;
  if (ln == 0) red[wv] = s;
  __syncthreads();
  if (threadIdx.x == 0) {
    float t = 0.0f;
    for (int w = 0; w < 8; ++w) t += red[w];
    dvec[row] = rsqrtf(1.0f + t);
  }
}

// Convert 8 f32 (two float4) into bf16 elements [base..base+7] of a fragment.
__device__ inline void cvt8_into(v16bf& dst, int base, f4 a, f4 b) {
  dst[base + 0] = (__bf16)a[0]; dst[base + 1] = (__bf16)a[1];
  dst[base + 2] = (__bf16)a[2]; dst[base + 3] = (__bf16)a[3];
  dst[base + 4] = (__bf16)b[0]; dst[base + 5] = (__bf16)b[1];
  dst[base + 6] = (__bf16)b[2]; dst[base + 7] = (__bf16)b[3];
}

// ---------------------------------------------------------------------------
// Kernel 2: hsT[c][j] = bf16( d_j * sum_k x[j,k]*W[c,k] ) via WMMA bf16.
// A-frag (16x32, 16-bit): lane l<16 -> row M=l, K=0..15 contiguous;
//                         lane l>=16 -> same row, K=16..31.
// B-frag (32x16): lane n<16 -> col n, K=0..15 contiguous; n>=16 -> K=16..31.
// ---------------------------------------------------------------------------
__global__ __launch_bounds__(256) void linear_hsT(const float* __restrict__ x,
                                                  const float* __restrict__ W,
                                                  const float* __restrict__ dvec,
                                                  __bf16* __restrict__ hsT) {
  const int wave = (blockIdx.x * blockDim.x + threadIdx.x) >> 5;
  const int lane = threadIdx.x & 31;
  const int cIdx = wave & 15;   // 16 tiles over DOUT
  const int jIdx = wave >> 4;   // 512 tiles over N
  const int c0 = cIdx * 16, j0 = jIdx * 16;
  const int r = lane & 15, hi = lane >> 4;

  const float* wrow = W + (size_t)(c0 + r) * DD;
  const float* xrow = x + (size_t)(j0 + r) * DD;
  const int akoff = hi * 16;
  const int bks = hi * 16;

  v8f acc = {};
  for (int k0 = 0; k0 < DD; k0 += 32) {
    f4 a0 = *(const f4*)(wrow + k0 + akoff);
    f4 a1 = *(const f4*)(wrow + k0 + akoff + 4);
    f4 a2 = *(const f4*)(wrow + k0 + akoff + 8);
    f4 a3 = *(const f4*)(wrow + k0 + akoff + 12);
    v16bf afrag;
    cvt8_into(afrag, 0, a0, a1);
    cvt8_into(afrag, 8, a2, a3);

    f4 b0 = *(const f4*)(xrow + k0 + bks);
    f4 b1 = *(const f4*)(xrow + k0 + bks + 4);
    f4 b2 = *(const f4*)(xrow + k0 + bks + 8);
    f4 b3 = *(const f4*)(xrow + k0 + bks + 12);
    v16bf bfrag;
    cvt8_into(bfrag, 0, b0, b1);
    cvt8_into(bfrag, 8, b2, b3);

    acc = __builtin_amdgcn_wmma_f32_16x16x32_bf16(false, afrag, false, bfrag,
                                                  (short)0, acc, false, false);
  }

  // C layout: VGPR rr, lanes 0-15 -> (M=rr, N=lane); lanes 16-31 -> (M=rr+8, N=lane-16)
  const float dj = dvec[j0 + r];
  #pragma unroll
  for (int rr = 0; rr < 8; ++rr) {
    const int c = c0 + rr + 8 * hi;
    hsT[(size_t)c * NN + j0 + r] = (__bf16)(dj * acc[rr]);
  }
}

// ---------------------------------------------------------------------------
// Kernel 3: main GEMM  acc = A @ hs  (M=8192, K=8192, N=256), bf16 WMMA,
// f32 accumulate; epilogue out[i,c] = relu(d_i*(acc + hsT[c][i])).
// Block = 8 waves: 128 rows x 64 cols. All waves share the same 64 B-columns,
// so the 64x32 bf16 B tile (4KB) is staged once per block per k-step into LDS
// with global_load_async_to_lds_b128 (ASYNCcnt, double-buffered); waves read
// fragments from LDS. The A stream (HBM roofline) stays on global_load_b128
// with f32->bf16 conversion in-register.
// ---------------------------------------------------------------------------
__global__ __launch_bounds__(256) void gemm_main(const float* __restrict__ A,
                                                 const __bf16* __restrict__ hsT,
                                                 const float* __restrict__ dvec,
                                                 float* __restrict__ out) {
  __shared__ __align__(128) __bf16 bsm[2][64 * 32];  // [buf][c_local*32 + k_local]

  const int tid  = threadIdx.x;
  const int wv   = tid >> 5;
  const int lane = tid & 31;
  const int wave = blockIdx.x * 8 + wv;   // 0..2047
  const int mIdx = wave & 511;            // 512 row tiles of 16
  const int nIdx = wave >> 9;             // 4 col groups of 64 (uniform per block)
  const int i0 = mIdx * 16;
  const int c0 = nIdx * 64;
  const int r = lane & 15, hi = lane >> 4;

  // Producer role: thread tid copies one 16B segment of the B tile.
  const int prow = tid >> 2;              // c_local 0..63
  const int pseg = tid & 3;               // 4 x 16B per row
  const unsigned pgoff0 = (unsigned)(((size_t)(c0 + prow) * NN + pseg * 8) * sizeof(__bf16));
  const unsigned plds0  = (unsigned)(uintptr_t)(&bsm[0][0]) +
                          (unsigned)((prow * 32 + pseg * 8) * sizeof(__bf16));
  const unsigned bufstride = (unsigned)(64 * 32 * sizeof(__bf16));

  const float* arow = A + (size_t)(i0 + r) * NN;
  const int akoff = hi * 16;

  v8f acc0 = {}, acc1 = {}, acc2 = {}, acc3 = {};

  // prologue: issue B tile for k0=0 into buffer 0
  asm volatile("global_load_async_to_lds_b128 %0, %1, %2"
               :: "v"(plds0), "v"(pgoff0), "s"(hsT) : "memory");

  for (int k0 = 0; k0 < NN; k0 += 32) {
    const int cur = (k0 >> 5) & 1;

    if (k0 + 32 < NN) {
      // issue next tile into the other buffer (reads of it finished at the
      // trailing barrier of the previous iteration), keep it in flight
      const unsigned lds  = plds0 + (unsigned)(1 - cur) * bufstride;
      const unsigned goff = pgoff0 + (unsigned)((k0 + 32) * sizeof(__bf16));
      asm volatile("global_load_async_to_lds_b128 %0, %1, %2"
                   :: "v"(lds), "v"(goff), "s"(hsT) : "memory");
      asm volatile("s_wait_asynccnt 0x1" ::: "memory");  // current tile landed
    } else {
      asm volatile("s_wait_asynccnt 0x0" ::: "memory");
    }
    __syncthreads();  // current B tile visible to all waves

    // A fragment: 16 rows x 32 k, f32 -> bf16 in-register
    __builtin_prefetch(arow + k0 + 32 + akoff, 0, 1);
    f4 a0 = *(const f4*)(arow + k0 + akoff);
    f4 a1 = *(const f4*)(arow + k0 + akoff + 4);
    f4 a2 = *(const f4*)(arow + k0 + akoff + 8);
    f4 a3 = *(const f4*)(arow + k0 + akoff + 12);
    v16bf afrag;
    cvt8_into(afrag, 0, a0, a1);
    cvt8_into(afrag, 8, a2, a3);

    // B fragments from LDS: lane n -> col c_local = t*16 + r, K contiguous
    const __bf16* bb = &bsm[cur][0];
    v16bf bf0 = *(const v16bf*)(bb + (0 * 16 + r) * 32 + hi * 16);
    v16bf bf1 = *(const v16bf*)(bb + (1 * 16 + r) * 32 + hi * 16);
    v16bf bf2 = *(const v16bf*)(bb + (2 * 16 + r) * 32 + hi * 16);
    v16bf bf3 = *(const v16bf*)(bb + (3 * 16 + r) * 32 + hi * 16);

    acc0 = __builtin_amdgcn_wmma_f32_16x16x32_bf16(false, afrag, false, bf0,
                                                   (short)0, acc0, false, false);
    acc1 = __builtin_amdgcn_wmma_f32_16x16x32_bf16(false, afrag, false, bf1,
                                                   (short)0, acc1, false, false);
    acc2 = __builtin_amdgcn_wmma_f32_16x16x32_bf16(false, afrag, false, bf2,
                                                   (short)0, acc2, false, false);
    acc3 = __builtin_amdgcn_wmma_f32_16x16x32_bf16(false, afrag, false, bf3,
                                                   (short)0, acc3, false, false);

    __syncthreads();  // all reads of this buffer done before it is rewritten
  }

  // epilogue: d scaling, self-loop term, relu
  f4 da = *(const f4*)(dvec + i0 + hi * 8);
  f4 db = *(const f4*)(dvec + i0 + hi * 8 + 4);
  v8f accs[4] = {acc0, acc1, acc2, acc3};
  #pragma unroll
  for (int t = 0; t < 4; ++t) {
    const int c = c0 + t * 16 + r;
    #pragma unroll
    for (int rr = 0; rr < 8; ++rr) {
      const int i = i0 + rr + 8 * hi;
      const float di = (rr < 4) ? da[rr] : db[rr - 4];
      const float self = (float)hsT[(size_t)c * NN + i];
      const float v = di * (accs[t][rr] + self);
      out[(size_t)i * DD + c] = v > 0.0f ? v : 0.0f;
    }
  }
}

// ---------------------------------------------------------------------------
extern "C" void kernel_launch(void* const* d_in, const int* in_sizes, int n_in,
                              void* d_out, int out_size, void* d_ws, size_t ws_size,
                              hipStream_t stream) {
  const float* x = (const float*)d_in[0];  // [8192,256]
  const float* A = (const float*)d_in[1];  // [8192,8192]
  const float* W = (const float*)d_in[2];  // [256,256]

  float* out  = (float*)d_out;                  // [8192,256]
  float* Aout = out + (size_t)NN * DD;          // [8192,8192] (tuple element 2)

  float*  dvec = (float*)d_ws;                                  // 32 KB
  __bf16* hsT  = (__bf16*)((char*)d_ws + 64 * 1024);            // 4 MB, [256][8192]

  degree_copy<<<NN, 256, 0, stream>>>(A, Aout, dvec);
  linear_hsT<<<(16 * 512) / 8, 256, 0, stream>>>(x, W, dvec, hsT);
  gemm_main<<<(512 * 4) / 8, 256, 0, stream>>>(A, hsT, dvec, out);
}
